// MultiHeadedAttention_42399917146938
// MI455X (gfx1250) — matrix operand order repaired and checked by reading, not compiled, not run on hardware
//
#include <hip/hip_runtime.h>
#include <hip/hip_bf16.h>
#include <cstdint>

// Problem constants (B, S, D, H) = (2, 2048, 1024, 16), DH = 64
#define B_  2
#define S_  2048
#define D_  1024
#define H_  16
#define DH_ 64

typedef __attribute__((ext_vector_type(16))) __bf16    v16bf;
typedef __attribute__((ext_vector_type(8)))  float     v8f;
typedef __attribute__((ext_vector_type(4)))  uint32_t  u32x4;

// ---------------------------------------------------------------------------
// Fragment loaders: bf16 planes are laid out so every WMMA fragment is two
// contiguous 16-byte loads per lane (no conversion VALU in the hot loop).
//
// A 16x32 bf16 layout (ISA 7.12.2): lane half selects K octet; per-lane
// elements 0..7  <-> k = kb + half*8  + {0..7}
// elements 8..15 <-> k = kb + 16 + half*8 + {0..7}
// B 32x16 layout: lane = column; same K mapping as A (0..7 <-> kb + half*8, 8..15 <-> kb + 16 + half*8)
// ---------------------------------------------------------------------------
union frag_u { u32x4 u[2]; v16bf v; };

__device__ __forceinline__ v16bf load_a_frag(const __bf16* __restrict__ row,
                                             int kb, int half) {
  frag_u x;
  x.u[0] = *reinterpret_cast<const u32x4*>(row + kb + half * 8);
  x.u[1] = *reinterpret_cast<const u32x4*>(row + kb + 16 + half * 8);
  return x.v;
}

__device__ __forceinline__ v16bf load_b_frag(const __bf16* __restrict__ row,
                                             int kb, int half) {
  frag_u x;
  x.u[0] = *reinterpret_cast<const u32x4*>(row + kb + half * 8);
  x.u[1] = *reinterpret_cast<const u32x4*>(row + kb + 16 + half * 8);
  return x.v;
}

// A-slot k offset (used only for the LDS P-tile relayout)
__device__ __forceinline__ int a_k_of_slot(int i, int half) {
  return ((i & 3) * 2) + ((i >> 2) * 16) + half * 8;
}

__device__ __forceinline__ void split_pair(float x0, float x1,
                                           v16bf& hi, v16bf& lo, int i) {
  __bf16 h0 = (__bf16)x0;
  __bf16 h1 = (__bf16)x1;
  hi[2 * i]     = h0;
  hi[2 * i + 1] = h1;
  lo[2 * i]     = (__bf16)(x0 - (float)h0);
  lo[2 * i + 1] = (__bf16)(x1 - (float)h1);
}

__device__ __forceinline__ v8f wmma_bf16(v16bf a, v16bf b, v8f c) {
  return __builtin_amdgcn_wmma_f32_16x16x32_bf16(
      false, a, false, b, (short)0, c, false, false);
}

// ---------------------------------------------------------------------------
// Elementwise f32 -> (bf16 hi, bf16 lo) split, vectorized x4.
// ---------------------------------------------------------------------------
__global__ void __launch_bounds__(256)
split_kernel(const float* __restrict__ x, __bf16* __restrict__ hi,
             __bf16* __restrict__ lo, int n4) {
  int idx = blockIdx.x * 256 + threadIdx.x;
  if (idx >= n4) return;
  float4 v = reinterpret_cast<const float4*>(x)[idx];
  __bf16 h[4], l[4];
  const float vv[4] = {v.x, v.y, v.z, v.w};
#pragma unroll
  for (int i = 0; i < 4; ++i) {
    h[i] = (__bf16)vv[i];
    l[i] = (__bf16)(vv[i] - (float)h[i]);
  }
  reinterpret_cast<uint2*>(hi)[idx] = *reinterpret_cast<uint2*>(h);
  reinterpret_cast<uint2*>(lo)[idx] = *reinterpret_cast<uint2*>(l);
}

// ---------------------------------------------------------------------------
// 16x64 register-blocked bf16x3 GEMM strip: one A hi/lo fragment pair is
// reused across 4 N-subtiles (12 WMMAs per 20 b128 loads per K=32 step).
//   out[m][n] = sum_k A[m][k] * W[n][k]   (W row-major, rows = n, offset n0)
// ---------------------------------------------------------------------------
__device__ void gemm16x64_bf16x3(const __bf16* __restrict__ Ahi,
                                 const __bf16* __restrict__ Alo, int lda,
                                 const __bf16* __restrict__ Whi,
                                 const __bf16* __restrict__ Wlo, int ldw,
                                 int K, v8f c[4]) {
  const int l = threadIdx.x & 31;
  const int half = l >> 4;
  const int lane16 = l & 15;
  const __bf16* arh = Ahi + (size_t)lane16 * lda;
  const __bf16* arl = Alo + (size_t)lane16 * lda;
  const __bf16* brh = Whi + (size_t)lane16 * ldw;  // +t*16*ldw folds to imm
  const __bf16* brl = Wlo + (size_t)lane16 * ldw;
  for (int kb = 0; kb < K; kb += 32) {
    v16bf ah = load_a_frag(arh, kb, half);
    v16bf al = load_a_frag(arl, kb, half);
#pragma unroll
    for (int t = 0; t < 4; ++t) {
      const size_t ro = (size_t)(t * 16) * ldw;
      v16bf bh = load_b_frag(brh + ro, kb, half);
      v16bf bl = load_b_frag(brl + ro, kb, half);
      c[t] = wmma_bf16(ah, bh, c[t]);
      c[t] = wmma_bf16(ah, bl, c[t]);
      c[t] = wmma_bf16(al, bh, c[t]);
    }
  }
}

// ---------------------------------------------------------------------------
// Input projection:  P = X @ W^T + b (scaled), epilogue re-splits to bf16
// hi/lo planes for the attention kernel.
//   MODE 0: out planes [B, H, S, DH]       (q, k)
//   MODE 1: out planes [B, H, DH, S]       (v, transposed for P.V B-fragments)
// ---------------------------------------------------------------------------
template <int MODE>
__global__ void __launch_bounds__(256)
proj_kernel(const __bf16* __restrict__ Xhi, const __bf16* __restrict__ Xlo,
            const __bf16* __restrict__ Whi, const __bf16* __restrict__ Wlo,
            const float* __restrict__ bias, float scale,
            __bf16* __restrict__ outHi, __bf16* __restrict__ outLo) {
  const int wave = threadIdx.x >> 5;
  const int l = threadIdx.x & 31;
  const int half = l >> 4, lane16 = l & 15;
  const int ntiles_n = D_ / 64;  // 16 strips of 64 columns
  const int tile = blockIdx.x * 8 + wave;
  const int m0 = (tile / ntiles_n) * 16;
  const int n0 = (tile % ntiles_n) * 64;

  v8f c[4] = {};
  gemm16x64_bf16x3(Xhi + (size_t)m0 * D_, Xlo + (size_t)m0 * D_, D_,
                   Whi + (size_t)n0 * D_, Wlo + (size_t)n0 * D_, D_, D_, c);

#pragma unroll
  for (int t = 0; t < 4; ++t) {
    const int n = n0 + t * 16 + lane16;
    const int h = n / DH_, d = n % DH_;
    const float bn = bias[n];
#pragma unroll
    for (int r = 0; r < 8; ++r) {
      const int m = m0 + r + 8 * half;  // C layout: vgpr r, lane half
      const int b = m / S_, s = m % S_;
      const float v = (c[t][r] + bn) * scale;
      const __bf16 vh = (__bf16)v;
      const __bf16 vl = (__bf16)(v - (float)vh);
      size_t o;
      if (MODE == 0)
        o = (((size_t)(b * H_ + h) * S_) + s) * DH_ + d;    // [B,H,S,DH]
      else
        o = (((size_t)(b * H_ + h) * DH_) + d) * S_ + s;    // [B,H,DH,S]
      outHi[o] = vh;
      outLo[o] = vl;
    }
  }
}

// ---------------------------------------------------------------------------
// Output projection:  out = ctx @ Wo^T + bo   (f32 [B*S, D] store)
// ---------------------------------------------------------------------------
__global__ void __launch_bounds__(256)
oproj_kernel(const __bf16* __restrict__ Xhi, const __bf16* __restrict__ Xlo,
             const __bf16* __restrict__ Whi, const __bf16* __restrict__ Wlo,
             const float* __restrict__ bias, float* __restrict__ out) {
  const int wave = threadIdx.x >> 5;
  const int l = threadIdx.x & 31;
  const int half = l >> 4, lane16 = l & 15;
  const int ntiles_n = D_ / 64;
  const int tile = blockIdx.x * 8 + wave;
  const int m0 = (tile / ntiles_n) * 16;
  const int n0 = (tile % ntiles_n) * 64;

  v8f c[4] = {};
  gemm16x64_bf16x3(Xhi + (size_t)m0 * D_, Xlo + (size_t)m0 * D_, D_,
                   Whi + (size_t)n0 * D_, Wlo + (size_t)n0 * D_, D_, D_, c);

#pragma unroll
  for (int t = 0; t < 4; ++t) {
    const float bn = bias[n0 + t * 16 + lane16];
#pragma unroll
    for (int r = 0; r < 8; ++r) {
      const int m = m0 + r + 8 * half;
      out[(size_t)m * D_ + n0 + t * 16 + lane16] = c[t][r] + bn;
    }
  }
}

// ---------------------------------------------------------------------------
// Attention: two-pass masked softmax attention, 16 query rows per wave.
// q/k planes: [B,H,S,DH] bf16 hi/lo;  v planes: [B,H,DH,S] bf16 hi/lo.
// Epilogue emits ctx as bf16 hi/lo planes in flat [B*S, D] for oproj.
// ---------------------------------------------------------------------------
__global__ void __launch_bounds__(256)
attn_kernel(const __bf16* __restrict__ qhi, const __bf16* __restrict__ qlo,
            const __bf16* __restrict__ khi, const __bf16* __restrict__ klo,
            const __bf16* __restrict__ vThi, const __bf16* __restrict__ vTlo,
            const uint8_t* __restrict__ mask,   // [B,S,S] bool
            __bf16* __restrict__ ctxHi, __bf16* __restrict__ ctxLo,
            float* __restrict__ top) {          // [B,S,S] head-0 attn
  __shared__ float pbuf[8][16][32];  // per-wave prob tile staging (16 KB)

  const int wave = threadIdx.x >> 5;
  const int l = threadIdx.x & 31;
  const int half = l >> 4, lane16 = l & 15;

  const int QCH = S_ / 128;
  const int bh = blockIdx.x / QCH;
  const int qc = blockIdx.x % QCH;
  const int b = bh / H_, h = bh % H_;
  const int q0 = qc * 128 + wave * 16;

  const __bf16* qbh = qhi + (size_t)bh * S_ * DH_;
  const __bf16* qbl = qlo + (size_t)bh * S_ * DH_;
  const __bf16* kbh = khi + (size_t)bh * S_ * DH_;
  const __bf16* kbl = klo + (size_t)bh * S_ * DH_;
  const __bf16* vbh = vThi + (size_t)bh * DH_ * S_;
  const __bf16* vbl = vTlo + (size_t)bh * DH_ * S_;
  const uint8_t* mbase = mask + (size_t)b * S_ * S_;

  // Q fragments (DH=64 -> two K=32 blocks), resident in VGPRs
  const __bf16* qrow_h = qbh + (size_t)(q0 + lane16) * DH_;
  const __bf16* qrow_l = qbl + (size_t)(q0 + lane16) * DH_;
  v16bf q_hi[2], q_lo[2];
#pragma unroll
  for (int kb = 0; kb < 2; ++kb) {
    q_hi[kb] = load_a_frag(qrow_h, kb * 32, half);
    q_lo[kb] = load_a_frag(qrow_l, kb * 32, half);
  }

  // Score tile for 16 keys starting at k0 (contraction over DH, bf16x3)
  auto score_tile = [&](int k0) -> v8f {
    const __bf16* krow_h = kbh + (size_t)(k0 + lane16) * DH_;
    const __bf16* krow_l = kbl + (size_t)(k0 + lane16) * DH_;
    v8f s = {};
#pragma unroll
    for (int kb = 0; kb < 2; ++kb) {
      v16bf kh = load_b_frag(krow_h, kb * 32, half);
      v16bf kl = load_b_frag(krow_l, kb * 32, half);
      s = wmma_bf16(q_hi[kb], kh, s);
      s = wmma_bf16(q_hi[kb], kl, s);
      s = wmma_bf16(q_lo[kb], kh, s);
    }
    return s;
  };

  // ---- Pass 1: per-lane online max / sum-exp ------------------------------
  float pm[8], ps[8];
#pragma unroll
  for (int r = 0; r < 8; ++r) { pm[r] = -3.0e38f; ps[r] = 0.0f; }

  for (int k0 = 0; k0 < S_; k0 += 16) {
    v8f s = score_tile(k0);
#pragma unroll
    for (int r = 0; r < 8; ++r) {
      const int q = q0 + r + 8 * half;
      const int k = k0 + lane16;
      const float sv = mbase[(size_t)q * S_ + k] ? -1.0e18f : s[r];
      const float mn = fmaxf(pm[r], sv);
      ps[r] = ps[r] * __expf(pm[r] - mn) + __expf(sv - mn);
      pm[r] = mn;
    }
  }
  // fold across the 16 lanes holding the same row (wave32-native)
#pragma unroll
  for (int r = 0; r < 8; ++r) {
#pragma unroll
    for (int msk = 1; msk < 16; msk <<= 1) {
      const float om = __shfl_xor(pm[r], msk, 32);
      const float os = __shfl_xor(ps[r], msk, 32);
      const float mn = fmaxf(pm[r], om);
      ps[r] = ps[r] * __expf(pm[r] - mn) + os * __expf(om - mn);
      pm[r] = mn;
    }
  }
  float inv[8];
#pragma unroll
  for (int r = 0; r < 8; ++r) inv[r] = (ps[r] > 0.0f) ? (1.0f / ps[r]) : 0.0f;

  // ---- Pass 2: probs -> LDS relayout -> P.V -------------------------------
  v8f acc[4] = {};

  for (int k0 = 0; k0 < S_; k0 += 32) {
#pragma unroll
    for (int sub = 0; sub < 2; ++sub) {
      const int kt = k0 + sub * 16;
      v8f s = score_tile(kt);
#pragma unroll
      for (int r = 0; r < 8; ++r) {
        const int m = r + 8 * half;
        const int q = q0 + m;
        const int k = kt + lane16;
        const float sv = mbase[(size_t)q * S_ + k] ? -1.0e18f : s[r];
        const float p = __expf(sv - pm[r]) * inv[r];
        pbuf[wave][m][sub * 16 + lane16] = p;  // C layout -> LDS
        if (h == 0) top[((size_t)b * S_ + q) * S_ + k] = p;
      }
    }
    // same-wave LDS RAW: order the ds stores before the relayout loads
    asm volatile("s_wait_dscnt 0x0" ::: "memory");

    // P as A fragment (16x32) from LDS, split to bf16 hi/lo (only in-loop cvt)
    v16bf phi, plo;
#pragma unroll
    for (int i = 0; i < 8; ++i) {
      const int kk = a_k_of_slot(i, half);
      float2 pv = *reinterpret_cast<const float2*>(&pbuf[wave][lane16][kk]);
      split_pair(pv.x, pv.y, phi, plo, i);
    }

    // ctx += P(16x32) x V(32x64): 4 N-tiles of 16; V^T rows are contiguous
#pragma unroll
    for (int t = 0; t < 4; ++t) {
      const __bf16* vrow_h = vbh + (size_t)(t * 16 + lane16) * S_;
      const __bf16* vrow_l = vbl + (size_t)(t * 16 + lane16) * S_;
      v16bf vh = load_b_frag(vrow_h, k0, half);
      v16bf vl = load_b_frag(vrow_l, k0, half);
      acc[t] = wmma_bf16(phi, vh, acc[t]);
      acc[t] = wmma_bf16(phi, vl, acc[t]);
      acc[t] = wmma_bf16(plo, vh, acc[t]);
    }
  }

  // ctx out: bf16 hi/lo planes, flat [B*S, D] with heads merged back
#pragma unroll
  for (int t = 0; t < 4; ++t)
#pragma unroll
    for (int r = 0; r < 8; ++r) {
      const int q = q0 + r + 8 * half;
      const int d = h * DH_ + t * 16 + lane16;
      const float v = acc[t][r];
      const __bf16 vh = (__bf16)v;
      const __bf16 vl = (__bf16)(v - (float)vh);
      const size_t o = ((size_t)b * S_ + q) * D_ + d;
      ctxHi[o] = vh;
      ctxLo[o] = vl;
    }
}

// ---------------------------------------------------------------------------
// Host-side orchestration (graph-capture safe: launches only, all on stream)
// ---------------------------------------------------------------------------
extern "C" void kernel_launch(void* const* d_in, const int* in_sizes, int n_in,
                              void* d_out, int out_size, void* d_ws,
                              size_t ws_size, hipStream_t stream) {
  (void)in_sizes; (void)n_in; (void)out_size; (void)ws_size;

  const float*   key   = (const float*)d_in[0];
  const float*   value = (const float*)d_in[1];
  const float*   query = (const float*)d_in[2];
  const uint8_t* mask  = (const uint8_t*)d_in[3];
  const float*   Wk    = (const float*)d_in[4];
  const float*   bk    = (const float*)d_in[5];
  const float*   Wv    = (const float*)d_in[6];
  const float*   bv    = (const float*)d_in[7];
  const float*   Wq    = (const float*)d_in[8];
  const float*   bq    = (const float*)d_in[9];
  const float*   Wo    = (const float*)d_in[10];
  const float*   bo    = (const float*)d_in[11];

  float* out = (float*)d_out;                 // [B,S,D] f32
  float* top = out + (size_t)B_ * S_ * D_;    // [B,S,S] head-0 attn

  // Workspace layout: bf16 planes (hi/lo pairs), 128 MB total
  const size_t NX = (size_t)B_ * S_ * D_;     // 4M elements (8 MB bf16)
  const size_t NW = (size_t)D_ * D_;          // 1M elements (2 MB bf16)
  __bf16* p = (__bf16*)d_ws;
  __bf16 *xkH = p,        *xkL = xkH + NX;    // split inputs
  __bf16 *xvH = xkL + NX, *xvL = xvH + NX;
  __bf16 *xqH = xvL + NX, *xqL = xqH + NX;
  __bf16 *wkH = xqL + NX, *wkL = wkH + NW;    // split weights
  __bf16 *wvH = wkL + NW, *wvL = wvH + NW;
  __bf16 *wqH = wvL + NW, *wqL = wqH + NW;
  __bf16 *woH = wqL + NW, *woL = woH + NW;
  __bf16 *kH  = woL + NW, *kL  = kH + NX;     // k  [B,H,S,DH]
  __bf16 *vH  = kL + NX,  *vL  = vH + NX;     // vT [B,H,DH,S]
  __bf16 *qH  = vL + NX,  *qL  = qH + NX;     // q  [B,H,S,DH], pre-scaled
  __bf16 *cH  = qL + NX,  *cL  = cH + NX;     // ctx flat [B*S, D]

  const dim3 blk(256);

  // 1) one-time f32 -> bf16 hi/lo splits (hoists all conversion VALU)
  const int gx = (int)(NX / 4 + 255) / 256;
  const int gw = (int)(NW / 4 + 255) / 256;
  split_kernel<<<gx, blk, 0, stream>>>(key,   xkH, xkL, (int)(NX / 4));
  split_kernel<<<gx, blk, 0, stream>>>(value, xvH, xvL, (int)(NX / 4));
  split_kernel<<<gx, blk, 0, stream>>>(query, xqH, xqL, (int)(NX / 4));
  split_kernel<<<gw, blk, 0, stream>>>(Wk, wkH, wkL, (int)(NW / 4));
  split_kernel<<<gw, blk, 0, stream>>>(Wv, wvH, wvL, (int)(NW / 4));
  split_kernel<<<gw, blk, 0, stream>>>(Wq, wqH, wqL, (int)(NW / 4));
  split_kernel<<<gw, blk, 0, stream>>>(Wo, woH, woL, (int)(NW / 4));

  // 2) projections (bf16x3 GEMM, 16x64 register-blocked, load-only inner loop)
  const int tiles = (B_ * S_ / 16) * (D_ / 64);  // 4096 wave-strips
  const int pgrid = tiles / 8;                   // 512 blocks
  proj_kernel<0><<<pgrid, blk, 0, stream>>>(xkH, xkL, wkH, wkL, bk, 1.0f, kH, kL);
  proj_kernel<1><<<pgrid, blk, 0, stream>>>(xvH, xvL, wvH, wvL, bv, 1.0f, vH, vL);
  proj_kernel<0><<<pgrid, blk, 0, stream>>>(xqH, xqL, wqH, wqL, bq, 0.125f, qH, qL);

  // 3) attention
  const int agrid = B_ * H_ * (S_ / 128);
  attn_kernel<<<agrid, blk, 0, stream>>>(qH, qL, kH, kL, vH, vL, mask,
                                         cH, cL, top);

  // 4) output projection
  oproj_kernel<<<pgrid, blk, 0, stream>>>(cH, cL, woH, woL, bo, out);
}
